// TemporalGNN_43353399886390
// MI455X (gfx1250) — matrix operand order repaired and checked by reading, not compiled, not run on hardware
//
#include <hip/hip_runtime.h>
#include <hip/hip_bf16.h>

typedef __attribute__((ext_vector_type(16))) _Float16 v16h;
typedef __attribute__((ext_vector_type(8)))  float    v8f;

#define NB   2
#define NT   8
#define NN   10000
#define CIN  64
#define COUT 32
#define NE   160000
#define NET  170000          // E + N self loops
#define NF   (NB*NT)         // 16 frames
#define NSEQ (NB*NN)         // 20000 GRU sequences
#define NEG_SLOPE 0.2f

// fragment table: 8 xlr frags [w(2)][kb(2)][ntile(2)] + 6 Wih + 6 Whh = 20
#define NFRAG      20
#define FRAG_XLR   0
#define FRAG_WIH   8
#define FRAG_WHH   14

// ---- float <-> order-preserving unsigned encoding (for atomic segment max)
__device__ __forceinline__ unsigned fenc(float x) {
  unsigned u = __float_as_uint(x);
  return (u & 0x80000000u) ? ~u : (u | 0x80000000u);
}
__device__ __forceinline__ float fdec(unsigned u) {
  return __uint_as_float((u & 0x80000000u) ? (u & 0x7fffffffu) : ~u);
}

__device__ __forceinline__ float sigmf(float x) { return 1.0f / (1.0f + __expf(-x)); }

// ------------------------------------------------------------------ zero
__global__ void k_zero(float* p, int n) {
  int i = blockIdx.x * blockDim.x + threadIdx.x;
  if (i < n) p[i] = 0.0f;
}

// ---- pack f32 weights into per-lane f16 WMMA B-fragments (done once) ----
// Fragment f, lane l -> v16h at frags[f*32 + l]; consumers load 32B/frag.
__global__ void k_pack_b(const float* __restrict__ Wl, const float* __restrict__ Wr,
                         const float* __restrict__ Wih, const float* __restrict__ Whh,
                         v16h* __restrict__ frags) {
  int lane = threadIdx.x & 31;
  int frag = blockIdx.x;                 // one block (32 lanes) per fragment
  int col = lane & 15, khalf = lane >> 4;
  v16h v;
  if (frag < FRAG_WIH) {
    // xlr: B[k][n] = W[k*32 + n0 + col], k = kb + khalf*16 + j
    int w = frag >> 2, kbi = (frag >> 1) & 1, nt = frag & 1;
    const float* W = w ? Wr : Wl;
    int kb = kbi * 32, n0 = nt * 16;
    #pragma unroll
    for (int j = 0; j < 16; ++j)
      v[j] = (_Float16)W[(kb + khalf * 16 + j) * COUT + n0 + col];
  } else {
    // gru: B[k][g] = W[(g16+col)*32 + k], k = khalf*16 + j  (W is [96][32])
    bool isih = frag < FRAG_WHH;
    int g = isih ? frag - FRAG_WIH : frag - FRAG_WHH;
    const float* W = isih ? Wih : Whh;
    #pragma unroll
    for (int j = 0; j < 16; ++j)
      v[j] = (_Float16)W[(g * 16 + col) * COUT + khalf * 16 + j];
  }
  frags[frag * 32 + lane] = v;
}

// --------------------------------------------- self-loop attr mean (accum)
__global__ void k_loop_accum(const int* __restrict__ dst,
                             const float* __restrict__ eattr,
                             float* loop_sum, float* cnt) {
  int e = blockIdx.x * blockDim.x + threadIdx.x;
  if (e >= NE) return;
  int d = dst[e];
  atomicAdd(&loop_sum[d * 3 + 0], eattr[e * 3 + 0]);
  atomicAdd(&loop_sum[d * 3 + 1], eattr[e * 3 + 1]);
  atomicAdd(&loop_sum[d * 3 + 2], eattr[e * 3 + 2]);
  atomicAdd(&cnt[d], 1.0f);
}

__global__ void k_loop_final(float* loop_sum, const float* cnt) {
  int n = blockIdx.x * blockDim.x + threadIdx.x;
  if (n >= NN) return;
  float inv = 1.0f / fmaxf(cnt[n], 1.0f);
  loop_sum[n * 3 + 0] *= inv;
  loop_sum[n * 3 + 1] *= inv;
  loop_sum[n * 3 + 2] *= inv;
}

// ------------------------------------------ edge embedding ee = attr @ W_e
__global__ void k_ee(const float* __restrict__ eattr,
                     const float* __restrict__ loop_attr,
                     const float* __restrict__ We, float* ee) {
  int e = blockIdx.x * blockDim.x + threadIdx.x;
  if (e >= NET) return;
  float a0, a1, a2;
  if (e < NE) { a0 = eattr[e*3+0]; a1 = eattr[e*3+1]; a2 = eattr[e*3+2]; }
  else { int n = e - NE; a0 = loop_attr[n*3+0]; a1 = loop_attr[n*3+1]; a2 = loop_attr[n*3+2]; }
  float4* eo = (float4*)(ee + e * COUT);
  #pragma unroll
  for (int q = 0; q < COUT / 4; ++q) {
    float4 w0 = ((const float4*)We)[q];
    float4 w1 = ((const float4*)(We + COUT))[q];
    float4 w2 = ((const float4*)(We + 2 * COUT))[q];
    float4 r;
    r.x = a0 * w0.x + a1 * w1.x + a2 * w2.x;
    r.y = a0 * w0.y + a1 * w1.y + a2 * w2.y;
    r.z = a0 * w0.z + a1 * w1.z + a2 * w2.z;
    r.w = a0 * w0.w + a1 * w1.w + a2 * w2.w;
    eo[q] = r;
  }
}

// ---------------- xl/xr = x @ W_l/W_r + b  via v_wmma_f32_16x16x32_f16 ----
// One wave -> 16 rows x 32 cols for BOTH xl and xr. Mtot = NF*NN = 160000.
__global__ void k_xlr_wmma(const float* __restrict__ x,
                           const v16h* __restrict__ frags,
                           const float* __restrict__ bl, const float* __restrict__ br,
                           float* __restrict__ xl, float* __restrict__ xr) {
  const int Mtot = NF * NN;
  int wave = (blockIdx.x * blockDim.x + threadIdx.x) >> 5;
  int lane = threadIdx.x & 31;
  int row0 = wave * 16;
  if (row0 >= Mtot) return;
  int row = lane & 15, khalf = lane >> 4, col = lane & 15;

  v8f al0 = {}, al1 = {}, ar0 = {}, ar1 = {};
  #pragma unroll
  for (int kbi = 0; kbi < 2; ++kbi) {
    int kb = kbi * 32;
    // A tile 16x32 (M x K): one contiguous 16-float run -> b128 loads
    v16h a;
    #pragma unroll
    for (int j = 0; j < 16; ++j) {
      int k = kb + khalf * 16 + j;
      a[j] = (_Float16)x[(row0 + row) * CIN + k];
    }
    // pre-packed B fragments: one 32B load each
    v16h b0 = frags[(FRAG_XLR + 0 + kbi * 2 + 0) * 32 + lane];   // Wl, n 0..15
    v16h b1 = frags[(FRAG_XLR + 0 + kbi * 2 + 1) * 32 + lane];   // Wl, n 16..31
    v16h b2 = frags[(FRAG_XLR + 4 + kbi * 2 + 0) * 32 + lane];   // Wr, n 0..15
    v16h b3 = frags[(FRAG_XLR + 4 + kbi * 2 + 1) * 32 + lane];   // Wr, n 16..31
    al0 = __builtin_amdgcn_wmma_f32_16x16x32_f16(false, a, false, b0, (short)0, al0, false, false);
    al1 = __builtin_amdgcn_wmma_f32_16x16x32_f16(false, a, false, b1, (short)0, al1, false, false);
    ar0 = __builtin_amdgcn_wmma_f32_16x16x32_f16(false, a, false, b2, (short)0, ar0, false, false);
    ar1 = __builtin_amdgcn_wmma_f32_16x16x32_f16(false, a, false, b3, (short)0, ar1, false, false);
  }
  #pragma unroll
  for (int r = 0; r < 8; ++r) {
    int orow = row0 + khalf * 8 + r;
    xl[orow * COUT + col]      = al0[r] + bl[col];
    xl[orow * COUT + col + 16] = al1[r] + bl[col + 16];
    xr[orow * COUT + col]      = ar0[r] + br[col];
    xr[orow * COUT + col + 16] = ar1[r] + br[col + 16];
  }
}

// ------------------------- attention logits + segment max over dst -------
__global__ void k_logit(const int* __restrict__ src, const int* __restrict__ dst,
                        const float* __restrict__ xl, const float* __restrict__ xr,
                        const float* __restrict__ ee, const float* __restrict__ att,
                        float* __restrict__ logitb, unsigned* __restrict__ emax) {
  int id = blockIdx.x * blockDim.x + threadIdx.x;
  if (id >= NF * NET) return;
  int f = id / NET, e = id - f * NET;
  int s = (e < NE) ? src[e] : (e - NE);
  int d = (e < NE) ? dst[e] : (e - NE);
  const float4* pa = (const float4*)(xl + (f * NN + s) * COUT);
  const float4* pb = (const float4*)(xr + (f * NN + d) * COUT);
  const float4* pe = (const float4*)(ee + e * COUT);
  const float4* pt = (const float4*)att;
  float sum = 0.0f;
  #pragma unroll
  for (int q = 0; q < COUT / 4; ++q) {
    float4 va = pa[q], vb = pb[q], ve = pe[q], vt = pt[q];
    float v;
    v = va.x + vb.x + ve.x; v = (v > 0.0f) ? v : NEG_SLOPE * v; sum += v * vt.x;
    v = va.y + vb.y + ve.y; v = (v > 0.0f) ? v : NEG_SLOPE * v; sum += v * vt.y;
    v = va.z + vb.z + ve.z; v = (v > 0.0f) ? v : NEG_SLOPE * v; sum += v * vt.z;
    v = va.w + vb.w + ve.w; v = (v > 0.0f) ? v : NEG_SLOPE * v; sum += v * vt.w;
  }
  logitb[id] = sum;
  atomicMax(&emax[f * NN + d], fenc(sum));
}

// ------------------------- softmax: exp(e - max), segment sum ------------
__global__ void k_expsum(const int* __restrict__ dst,
                         float* __restrict__ logitb,
                         const unsigned* __restrict__ emax,
                         float* __restrict__ denom) {
  int id = blockIdx.x * blockDim.x + threadIdx.x;
  if (id >= NF * NET) return;
  int f = id / NET, e = id - f * NET;
  int d = (e < NE) ? dst[e] : (e - NE);
  float w = __expf(logitb[id] - fdec(emax[f * NN + d]));
  logitb[id] = w;
  atomicAdd(&denom[f * NN + d], w);
}

// ---------- aggregate: acc[seq][t][c] += alpha * xl[src] (GRU layout) ----
__global__ void k_agg(const int* __restrict__ src, const int* __restrict__ dst,
                      const float* __restrict__ logitb, const float* __restrict__ denom,
                      const float* __restrict__ xl, float* __restrict__ accb) {
  int id = blockIdx.x * blockDim.x + threadIdx.x;
  if (id >= NF * NET) return;
  int f = id / NET, e = id - f * NET;
  int s = (e < NE) ? src[e] : (e - NE);
  int d = (e < NE) ? dst[e] : (e - NE);
  float coef = logitb[id] / denom[f * NN + d];
  int b = f / NT, t = f - b * NT;
  int base = ((b * NN + d) * NT + t) * COUT;      // [seq][t][c]
  const float4* pl = (const float4*)(xl + (f * NN + s) * COUT);
  #pragma unroll
  for (int q = 0; q < COUT / 4; ++q) {
    float4 v = pl[q];
    atomicAdd(&accb[base + q * 4 + 0], coef * v.x);
    atomicAdd(&accb[base + q * 4 + 1], coef * v.y);
    atomicAdd(&accb[base + q * 4 + 2], coef * v.z);
    atomicAdd(&accb[base + q * 4 + 3], coef * v.w);
  }
}

// ------------------- GRU over T=8, 16 sequences per wave, WMMA gates -----
__global__ void k_gru_wmma(const float* __restrict__ accb, const float* __restrict__ bias_gat,
                           const v16h* __restrict__ frags,
                           const float* __restrict__ bih, const float* __restrict__ bhh,
                           float* __restrict__ out) {
  __shared__ float ldsh[8][16][33];                // per-wave 16x32 h tile (+pad)
  int wslot = threadIdx.x >> 5;
  int lane  = threadIdx.x & 31;
  int wave  = (blockIdx.x * blockDim.x + threadIdx.x) >> 5;
  int seq0  = wave * 16;
  if (seq0 >= NSEQ) return;
  int row = lane & 15, khalf = lane >> 4, col = lane & 15;

  // h0 = 0 (registers in C-layout + LDS copy in row-major)
  float hC[2][8];
  #pragma unroll
  for (int r = 0; r < 8; ++r) {
    hC[0][r] = 0.0f; hC[1][r] = 0.0f;
    ldsh[wslot][khalf * 8 + r][col]      = 0.0f;
    ldsh[wslot][khalf * 8 + r][col + 16] = 0.0f;
  }
  asm volatile("s_wait_dscnt 0" ::: "memory");

  for (int t = 0; t < NT; ++t) {
    // A matrices: gi input (GAT output + bias_gat) and previous hidden state
    v16h ain, ah;
    #pragma unroll
    for (int j = 0; j < 16; ++j) {
      int k = khalf * 16 + j;
      int seq = seq0 + row;
      ain[j] = (_Float16)(accb[(seq * NT + t) * COUT + k] + bias_gat[k]);
      ah[j]  = (_Float16)ldsh[wslot][row][k];
    }
    // gi = h_in @ W_ih^T  (6 tiles of 16 cols), gh = h @ W_hh^T
    v8f gi[6], gh[6];
    #pragma unroll
    for (int g = 0; g < 6; ++g) {
      v16h bt = frags[(FRAG_WIH + g) * 32 + lane];
      v8f z = {};
      gi[g] = __builtin_amdgcn_wmma_f32_16x16x32_f16(false, ain, false, bt, (short)0, z, false, false);
    }
    #pragma unroll
    for (int g = 0; g < 6; ++g) {
      v16h bt = frags[(FRAG_WHH + g) * 32 + lane];
      v8f z = {};
      gh[g] = __builtin_amdgcn_wmma_f32_16x16x32_f16(false, ah, false, bt, (short)0, z, false, false);
    }
    // gates (torch order r,z,n) and state update
    #pragma unroll
    for (int half = 0; half < 2; ++half) {
      int cc = half * 16 + col;
      float bi_r = bih[cc],      bh_r = bhh[cc];
      float bi_z = bih[32 + cc], bh_z = bhh[32 + cc];
      float bi_n = bih[64 + cc], bh_n = bhh[64 + cc];
      #pragma unroll
      for (int r = 0; r < 8; ++r) {
        float rg = sigmf(gi[half][r]     + bi_r + gh[half][r]     + bh_r);
        float zg = sigmf(gi[2 + half][r] + bi_z + gh[2 + half][r] + bh_z);
        float ng = tanhf(gi[4 + half][r] + bi_n + rg * (gh[4 + half][r] + bh_n));
        float hv = (1.0f - zg) * ng + zg * hC[half][r];
        hC[half][r] = hv;
        int mrow = khalf * 8 + r;
        int seq  = seq0 + mrow;
        int b = seq / NN, n = seq - b * NN;
        out[(((b * NT + t) * NN) + n) * COUT + cc] = hv;   // [B,T,N,C]
        ldsh[wslot][mrow][cc] = hv;
      }
    }
    asm volatile("s_wait_dscnt 0" ::: "memory");
  }
}

// ========================================================================
extern "C" void kernel_launch(void* const* d_in, const int* in_sizes, int n_in,
                              void* d_out, int out_size, void* d_ws, size_t ws_size,
                              hipStream_t stream) {
  const float* x     = (const float*)d_in[0];
  const int*   eidx  = (const int*)d_in[1];
  const float* eattr = (const float*)d_in[2];
  const float* Wl    = (const float*)d_in[3];
  const float* bl    = (const float*)d_in[4];
  const float* Wr    = (const float*)d_in[5];
  const float* br    = (const float*)d_in[6];
  const float* We    = (const float*)d_in[7];
  const float* att   = (const float*)d_in[8];
  const float* bgat  = (const float*)d_in[9];
  const float* Wih   = (const float*)d_in[10];
  const float* Whh   = (const float*)d_in[11];
  const float* bih   = (const float*)d_in[12];
  const float* bhh   = (const float*)d_in[13];
  const int* srcp = eidx;
  const int* dstp = eidx + NE;
  float* out = (float*)d_out;

  // workspace layout (floats); frags first for 32B alignment
  float* ws = (float*)d_ws;
  size_t o = 0;
  v16h*  frags    = (v16h*)ws;  o += (size_t)NFRAG * 32 * 8;   // 20KB of f16 frags
  float* loop_sum = ws + o; o += (size_t)NN * 3;
  float* cnt      = ws + o; o += NN;
  float* ee       = ws + o; o += (size_t)NET * COUT;
  float* xl       = ws + o; o += (size_t)NF * NN * COUT;
  float* xr       = ws + o; o += (size_t)NF * NN * COUT;
  float* logitb   = ws + o; o += (size_t)NF * NET;
  float* emax     = ws + o; o += (size_t)NF * NN;
  float* denom    = ws + o; o += (size_t)NF * NN;
  float* accb     = ws + o; o += (size_t)NSEQ * NT * COUT;

  const int TB = 256;
  auto blk = [](int n, int tb) { return (n + tb - 1) / tb; };

  // deterministic init each launch
  k_zero<<<blk(NN * 4, TB), TB, 0, stream>>>(loop_sum, NN * 4);           // loop_sum + cnt
  k_zero<<<blk(NF * NN, TB), TB, 0, stream>>>(emax, NF * NN);             // 0 < fenc(-inf): identity for atomicMax
  k_zero<<<blk(NF * NN, TB), TB, 0, stream>>>(denom, NF * NN);
  k_zero<<<blk(NSEQ * NT * COUT, TB), TB, 0, stream>>>(accb, NSEQ * NT * COUT);

  // pack weight fragments (f32 -> f16, WMMA per-lane layout)
  k_pack_b<<<NFRAG, 32, 0, stream>>>(Wl, Wr, Wih, Whh, frags);

  // self-loop attr mean, edge embeddings
  k_loop_accum<<<blk(NE, TB), TB, 0, stream>>>(dstp, eattr, loop_sum, cnt);
  k_loop_final<<<blk(NN, TB), TB, 0, stream>>>(loop_sum, cnt);
  k_ee<<<blk(NET, TB), TB, 0, stream>>>(eattr, loop_sum, We, ee);

  // node transforms (WMMA f16->f32): 160000 rows, 16 rows/wave, 8 waves/block
  k_xlr_wmma<<<(NF * NN / 16) / 8, TB, 0, stream>>>(x, frags, bl, br, xl, xr);

  // GATv2 attention + softmax + aggregation
  int nfe = NF * NET;
  k_logit<<<blk(nfe, TB), TB, 0, stream>>>(srcp, dstp, xl, xr, ee, att,
                                           logitb, (unsigned*)emax);
  k_expsum<<<blk(nfe, TB), TB, 0, stream>>>(dstp, logitb, (const unsigned*)emax, denom);
  k_agg<<<blk(nfe, TB), TB, 0, stream>>>(srcp, dstp, logitb, denom, xl, accb);

  // GRU scan (WMMA gates): 20000 seqs / 16 per wave = 1250 waves
  k_gru_wmma<<<blk(NSEQ / 16, 8), TB, 0, stream>>>(accb, bgat, frags, bih, bhh, out);
}
